// EquivariantSelfAttention_25890062860710
// MI455X (gfx1250) — matrix operand order repaired and checked by reading, not compiled, not run on hardware
//
#include <hip/hip_runtime.h>

typedef __attribute__((ext_vector_type(16))) _Float16 v16h;
typedef __attribute__((ext_vector_type(8)))  _Float16 v8h;
typedef __attribute__((ext_vector_type(8)))  float    v8f;
typedef __attribute__((ext_vector_type(2)))  float    v2f;

namespace {
constexpr int kN = 1536, kC0 = 128, kC1 = 64, kH = 8, kNRBF = 16;
constexpr int QK_LD = 64;    // padded per-head q/k feature dim (40 used)
constexpr int V_LD  = 48;    // padded per-head v feature dim (40 used)
// folded path norms: score = (qs.ks/128 + qv.kv/64)/sqrt(192)
constexpr float S_QS = 6.3789106e-3f;   // (1/sqrt(128)) * (1/sqrt(192))
constexpr float S_KS = 8.8388348e-2f;   // 1/sqrt(128)
constexpr float S_QV = 9.0210999e-3f;   // (1/8) * (1/sqrt(192))
constexpr float S_KV = 0.125f;
constexpr float S_VS = 8.8388348e-2f;
constexpr float S_VV = 0.125f;
constexpr float S_OUT_S = 8.5749293e-2f; // 1/sqrt(136)
constexpr float S_OUT_V = 1.1785113e-1f; // 1/sqrt(72)
constexpr float SQRT3   = 1.7320508f;
}

__device__ inline v8f wmma_f16(v16h a, v16h b, v8f c) {
  return __builtin_amdgcn_wmma_f32_16x16x32_f16(false, a, false, b, (short)0, c,
                                                false, false);
}
__device__ inline v8f wmma_f32(v2f a, v2f b, v8f c) {
  return __builtin_amdgcn_wmma_f32_16x16x4_f32(false, a, false, b, (short)0, c,
                                               false, false);
}
__device__ inline v16h cat8(v8h lo, v8h hi) {
  return __builtin_shufflevector(lo, hi, 0, 1, 2, 3, 4, 5, 6, 7, 8, 9, 10, 11,
                                 12, 13, 14, 15);
}
__device__ inline float redmax16(float t) {
  t = fmaxf(t, __shfl_xor(t, 1));
  t = fmaxf(t, __shfl_xor(t, 2));
  t = fmaxf(t, __shfl_xor(t, 4));
  t = fmaxf(t, __shfl_xor(t, 8));
  return t;
}
__device__ inline float redsum16(float t) {
  t += __shfl_xor(t, 1);
  t += __shfl_xor(t, 2);
  t += __shfl_xor(t, 4);
  t += __shfl_xor(t, 8);
  return t;
}

// ---- projections: scalar channels (x_s @ {wq_s,wk_s,wv_s}), packed per head
__global__ void __launch_bounds__(32)
proj_s_kernel(const float* __restrict__ x_s, const float* __restrict__ wq,
              const float* __restrict__ wk, const float* __restrict__ wv,
              _Float16* __restrict__ Qp, _Float16* __restrict__ Kp,
              _Float16* __restrict__ Vt) {
  const int lane = threadIdx.x, hl = lane >> 4, n = lane & 15;
  const int rt = blockIdx.x, mat = blockIdx.y >> 3, ct = blockIdx.y & 7;
  const float* W = (mat == 0) ? wq : (mat == 1) ? wk : wv;
  const int row = rt * 16 + n, col = ct * 16 + n;
  v8f c = {};
  for (int k0 = 0; k0 < kC0; k0 += 4) {
    int k = k0 + hl * 2;
    v2f a = {x_s[row * kC0 + k], x_s[row * kC0 + k + 1]};
    v2f b = {W[k * kC0 + col], W[(k + 1) * kC0 + col]};
    c = wmma_f32(a, b, c);
  }
  const int h = col >> 4, cc = col & 15;
#pragma unroll
  for (int v = 0; v < 8; ++v) {
    int i = rt * 16 + v + 8 * hl;
    float val = c[v];
    if (mat == 0)      Qp[(h * kN + i) * QK_LD + cc] = (_Float16)(val * S_QS);
    else if (mat == 1) Kp[(h * kN + i) * QK_LD + cc] = (_Float16)(val * S_KS);
    else               Vt[(h * V_LD + cc) * kN + i]  = (_Float16)(val * S_VS);
  }
}

// ---- projections: vector channels (einsum ncd,ce->ned per axis d)
__global__ void __launch_bounds__(32)
proj_v_kernel(const float* __restrict__ x_v, const float* __restrict__ wq,
              const float* __restrict__ wk, const float* __restrict__ wv,
              _Float16* __restrict__ Qp, _Float16* __restrict__ Kp,
              _Float16* __restrict__ Vt) {
  const int lane = threadIdx.x, hl = lane >> 4, n = lane & 15;
  const int rt = blockIdx.x, mat = blockIdx.y / 12;
  const int rem = blockIdx.y % 12, d = rem >> 2, ct = rem & 3;
  const float* W = (mat == 0) ? wq : (mat == 1) ? wk : wv;
  const int row = rt * 16 + n, col = ct * 16 + n;
  v8f c = {};
  for (int k0 = 0; k0 < kC1; k0 += 4) {
    int k = k0 + hl * 2;
    v2f a = {x_v[(row * kC1 + k) * 3 + d], x_v[(row * kC1 + k + 1) * 3 + d]};
    v2f b = {W[k * kC1 + col], W[(k + 1) * kC1 + col]};
    c = wmma_f32(a, b, c);
  }
  const int h = col >> 3, cc = col & 7, pos = 16 + cc * 3 + d;
#pragma unroll
  for (int v = 0; v < 8; ++v) {
    int i = rt * 16 + v + 8 * hl;
    float val = c[v];
    if (mat == 0)      Qp[(h * kN + i) * QK_LD + pos] = (_Float16)(val * S_QV);
    else if (mat == 1) Kp[(h * kN + i) * QK_LD + pos] = (_Float16)(val * S_KV);
    else               Vt[(h * V_LD + pos) * kN + i]  = (_Float16)(val * S_VV);
  }
}

// ---- fused attention: scores (WMMA) + RBF/MLP bias + online softmax + P@V (WMMA)
__global__ void __launch_bounds__(32)
attn_kernel(const _Float16* __restrict__ Qp, const _Float16* __restrict__ Kp,
            const _Float16* __restrict__ Vt, const float* __restrict__ coord,
            const float* __restrict__ w_ang_s, const float* __restrict__ w_ang_v,
            const float* __restrict__ w_mlp, const float* __restrict__ b_mlp,
            float* __restrict__ MS, float* __restrict__ MVd) {
  __shared__ __align__(16) _Float16 plds[16 * 32];
  const int lane = threadIdx.x, hl = lane >> 4, n = lane & 15;
  const int q0 = blockIdx.x * 16, h = blockIdx.y;

  // Q in A-matrix layout, two K-chunks (k 0..31 and 32..63)
  const _Float16* qrow = Qp + (size_t)(h * kN + q0 + n) * QK_LD;
  v16h qa0 = *(const v16h*)(qrow + hl * 16);
  v16h qa1 = *(const v16h*)(qrow + 32 + hl * 16);

  float cix[8], ciy[8], ciz[8];
#pragma unroll
  for (int v = 0; v < 8; ++v) {
    int i = q0 + v + 8 * hl;
    cix[v] = coord[i * 3 + 0];
    ciy[v] = coord[i * 3 + 1];
    ciz[v] = coord[i * 3 + 2];
  }
  float wm[kNRBF];
#pragma unroll
  for (int k = 0; k < kNRBF; ++k) wm[k] = w_mlp[k * kH + h];
  const float bm = b_mlp[h], was = w_ang_s[h], wav = w_ang_v[h];

  v8f acc0 = {}, acc1 = {}, acc2 = {};
  float Mrow[8], lpart[8], sux[8], suy[8], suz[8];
#pragma unroll
  for (int v = 0; v < 8; ++v) {
    Mrow[v] = -3.0e38f; lpart[v] = 0.f; sux[v] = 0.f; suy[v] = 0.f; suz[v] = 0.f;
  }

  for (int j0 = 0; j0 < kN; j0 += 32) {
    // K in B-matrix layout: lane = key column, contiguous 16 k-halves
    const _Float16* kr0 = Kp + (size_t)(h * kN + j0 + n) * QK_LD;
    const _Float16* kr1 = Kp + (size_t)(h * kN + j0 + 16 + n) * QK_LD;
    v16h kb00 = *(const v16h*)(kr0 + hl * 16);
    v16h kb01 = *(const v16h*)(kr0 + 32 + hl * 16);
    v16h kb10 = *(const v16h*)(kr1 + hl * 16);
    v16h kb11 = *(const v16h*)(kr1 + 32 + hl * 16);
    v8f s0 = {}; s0 = wmma_f16(qa0, kb00, s0); s0 = wmma_f16(qa1, kb01, s0);
    v8f s1 = {}; s1 = wmma_f16(qa0, kb10, s1); s1 = wmma_f16(qa1, kb11, s1);

    // geometric bias per element (C layout: fixed key column per lane)
    float sc[2][8], ux[2][8], uy[2][8], uz[2][8];
#pragma unroll
    for (int jt = 0; jt < 2; ++jt) {
      int j = j0 + jt * 16 + n;
      float cjx = coord[j * 3 + 0], cjy = coord[j * 3 + 1], cjz = coord[j * 3 + 2];
#pragma unroll
      for (int v = 0; v < 8; ++v) {
        float rx = cix[v] - cjx, ry = ciy[v] - cjy, rz = ciz[v] - cjz;
        float n2 = rx * rx + ry * ry + rz * rz;
        float dist = 0.f, invd = 0.f;
        if (n2 > 0.f) { dist = sqrtf(n2); invd = 1.f / dist; }
        ux[jt][v] = rx * invd; uy[jt][v] = ry * invd; uz[jt][v] = rz * invd;
        float a = bm;
#pragma unroll
        for (int k = 0; k < kNRBF; ++k) {
          float t = (dist - (float)k * (1.f / 3.f)) * 2.0f;  // (d-c)/0.5
          a += __expf(-t * t) * wm[k];
        }
        float bias = a / (1.f + __expf(-a));                 // silu
        sc[jt][v] = ((jt == 0) ? s0[v] : s1[v]) + bias;
      }
    }

    // online softmax over these 32 keys; rescale running accumulators
#pragma unroll
    for (int v = 0; v < 8; ++v) {
      float t = redmax16(fmaxf(sc[0][v], sc[1][v]));
      float newM = fmaxf(Mrow[v], t);
      float scale = __expf(Mrow[v] - newM);
      Mrow[v] = newM;
      lpart[v] *= scale; sux[v] *= scale; suy[v] *= scale; suz[v] *= scale;
      acc0[v] *= scale; acc1[v] *= scale; acc2[v] *= scale;
      float p0 = __expf(sc[0][v] - newM);
      float p1 = __expf(sc[1][v] - newM);
      lpart[v] += p0 + p1;
      sux[v] += p0 * ux[0][v] + p1 * ux[1][v];
      suy[v] += p0 * uy[0][v] + p1 * uy[1][v];
      suz[v] += p0 * uz[0][v] + p1 * uz[1][v];
      int m = v + 8 * hl;
      plds[m * 32 + n] = (_Float16)p0;
      plds[m * 32 + 16 + n] = (_Float16)p1;
    }
    __syncthreads();
    // P back in A layout; V tiles in B layout (channel-major -> contiguous)
    v16h pa = cat8(*(const v8h*)(plds + n * 32 + hl * 16),
                   *(const v8h*)(plds + n * 32 + hl * 16 + 8));
    const _Float16* vb = Vt + (size_t)(h * V_LD + n) * kN + j0 + hl * 16;
    v16h vb0 = *(const v16h*)(vb);
    v16h vb1 = *(const v16h*)(vb + 16 * kN);
    v16h vb2 = *(const v16h*)(vb + 32 * kN);
    acc0 = wmma_f16(pa, vb0, acc0);
    acc1 = wmma_f16(pa, vb1, acc1);
    acc2 = wmma_f16(pa, vb2, acc2);
    __syncthreads();
  }

  // finalize: normalize by softmax denom, emit message tensors
#pragma unroll
  for (int v = 0; v < 8; ++v) {
    float l = redsum16(lpart[v]);
    float sx = redsum16(sux[v]);
    float sy = redsum16(suy[v]);
    float sz = redsum16(suz[v]);
    float invl = (l > 0.f) ? 1.f / l : 0.f;
    int i = q0 + v + 8 * hl;
    MS[i * 136 + h * 17 + n] = acc0[v] * invl;
    {  // value-channel rows 16..31 -> (c,d) of msg_v
      int c = n / 3, d = n % 3;
      MVd[((size_t)d * kN + i) * 72 + h * 9 + c] = acc1[v] * invl;
    }
    if (n < 8) {  // rows 32..39
      int rel = 16 + n, c = rel / 3, d = rel % 3;
      MVd[((size_t)d * kN + i) * 72 + h * 9 + c] = acc2[v] * invl;
    }
    if (n == 0) {
      MS[i * 136 + h * 17 + 16] = was * (l * invl);
      MVd[((size_t)0 * kN + i) * 72 + h * 9 + 8] = SQRT3 * wav * sx * invl;
      MVd[((size_t)1 * kN + i) * 72 + h * 9 + 8] = SQRT3 * wav * sy * invl;
      MVd[((size_t)2 * kN + i) * 72 + h * 9 + 8] = SQRT3 * wav * sz * invl;
    }
  }
}

// ---- output linear, scalar irreps: (N,136)@(136,128)
__global__ void __launch_bounds__(32)
out_s_kernel(const float* __restrict__ MS, const float* __restrict__ Wo,
             float* __restrict__ out) {
  const int lane = threadIdx.x, hl = lane >> 4, n = lane & 15;
  const int rt = blockIdx.x, ct = blockIdx.y;
  const int row = rt * 16 + n, col = ct * 16 + n;
  v8f c = {};
  for (int k0 = 0; k0 < 136; k0 += 4) {
    int k = k0 + hl * 2;
    v2f a = {MS[row * 136 + k], MS[row * 136 + k + 1]};
    v2f b = {Wo[k * kC0 + col], Wo[(k + 1) * kC0 + col]};
    c = wmma_f32(a, b, c);
  }
#pragma unroll
  for (int v = 0; v < 8; ++v) {
    int i = rt * 16 + v + 8 * hl;
    out[i * 320 + col] = c[v] * S_OUT_S;
  }
}

// ---- output linear, vector irreps per axis: (N,72)@(72,64)
__global__ void __launch_bounds__(32)
out_v_kernel(const float* __restrict__ MVd, const float* __restrict__ Wo,
             float* __restrict__ out) {
  const int lane = threadIdx.x, hl = lane >> 4, n = lane & 15;
  const int rt = blockIdx.x, ct = blockIdx.y & 3, d = blockIdx.y >> 2;
  const int row = rt * 16 + n, col = ct * 16 + n;
  const float* A = MVd + (size_t)d * kN * 72;
  v8f c = {};
  for (int k0 = 0; k0 < 72; k0 += 4) {
    int k = k0 + hl * 2;
    v2f a = {A[row * 72 + k], A[row * 72 + k + 1]};
    v2f b = {Wo[k * kC1 + col], Wo[(k + 1) * kC1 + col]};
    c = wmma_f32(a, b, c);
  }
#pragma unroll
  for (int v = 0; v < 8; ++v) {
    int i = rt * 16 + v + 8 * hl;
    out[i * 320 + 128 + col * 3 + d] = c[v] * S_OUT_V;
  }
}

extern "C" void kernel_launch(void* const* d_in, const int* in_sizes, int n_in,
                              void* d_out, int out_size, void* d_ws, size_t ws_size,
                              hipStream_t stream) {
  const float* x_s     = (const float*)d_in[0];
  const float* x_v     = (const float*)d_in[1];
  const float* coord   = (const float*)d_in[2];
  // d_in[3] = mask_coord (all true in this configuration)
  const float* wq_s    = (const float*)d_in[4];
  const float* wq_v    = (const float*)d_in[5];
  const float* wk_s    = (const float*)d_in[6];
  const float* wk_v    = (const float*)d_in[7];
  const float* wv_s    = (const float*)d_in[8];
  const float* wv_v    = (const float*)d_in[9];
  const float* w_ang_s = (const float*)d_in[10];
  const float* w_ang_v = (const float*)d_in[11];
  const float* w_mlp   = (const float*)d_in[12];
  const float* b_mlp   = (const float*)d_in[13];
  const float* w_out_s = (const float*)d_in[14];
  const float* w_out_v = (const float*)d_in[15];
  float* out = (float*)d_out;

  char* ws = (char*)d_ws;
  _Float16* Qp = (_Float16*)ws;
  _Float16* Kp = Qp + (size_t)kH * kN * QK_LD;
  _Float16* Vt = Kp + (size_t)kH * kN * QK_LD;
  const size_t halfBytes =
      ((size_t)2 * kH * kN * QK_LD + (size_t)kH * V_LD * kN) * sizeof(_Float16);
  float* MS  = (float*)(ws + halfBytes);
  float* MVd = MS + (size_t)kN * 136;

  // zero the padded f16 pack region (cols 40..63 / rows 40..47 stay zero)
  hipMemsetAsync(d_ws, 0, halfBytes, stream);

  proj_s_kernel<<<dim3(kN / 16, 24), 32, 0, stream>>>(x_s, wq_s, wk_s, wv_s,
                                                      Qp, Kp, Vt);
  proj_v_kernel<<<dim3(kN / 16, 36), 32, 0, stream>>>(x_v, wq_v, wk_v, wv_v,
                                                      Qp, Kp, Vt);
  attn_kernel<<<dim3(kN / 16, kH), 32, 0, stream>>>(Qp, Kp, Vt, coord, w_ang_s,
                                                    w_ang_v, w_mlp, b_mlp, MS, MVd);
  out_s_kernel<<<dim3(kN / 16, 8), 32, 0, stream>>>(MS, w_out_s, out);
  out_v_kernel<<<dim3(kN / 16, 12), 32, 0, stream>>>(MVd, w_out_v, out);
}
